// HRFormer_module_84138409328712
// MI455X (gfx1250) — compile-verified, matrix operand order checked by reading, not run on hardware
//
#include <hip/hip_runtime.h>
#include <hip/hip_bf16.h>
#include <math.h>

// ---------------------------------------------------------------------------
// HRFormer block (windowed MHSA + DW-CNN FFN) for MI455X / gfx1250.
// GEMMs: v_wmma_f32_16x16x32_f16 with double-buffered async weight staging
// (global_load_async_to_lds_b128 + s_wait_asynccnt).
// ---------------------------------------------------------------------------

typedef __attribute__((ext_vector_type(16))) _Float16 v16h;
typedef __attribute__((ext_vector_type(8)))  float    v8f;

#define BATCH 4
#define CH    256
#define HH_   112
#define WW_   112
#define HW_   (HH_*WW_)          // 12544
#define NTOK  (BATCH*HW_)        // 50176
#define WS    7
#define NH    8
#define HD    32                 // head dim
#define HID   1024
#define NWIN  1024               // B * 16 * 16
#define WPAD  64                 // padded window rows (49 -> 64)
#define MWIN  (NWIN*WPAD)        // 65536

#define LN_EPS 1e-6f
#define BN_EPS 1e-5f

__device__ __forceinline__ float gelu_exact(float x) {
    return 0.5f * x * (1.0f + erff(x * 0.7071067811865475f));
}

// WMMA 16-bit fragment load (A or B) from row-major (rows x K).
// p = base + row*ld + kb, row = lane&15, kb = (lane>>4)*8.
// halves 0..7 <- K kb..kb+7 ; halves 8..15 <- K kb+16..kb+23.
__device__ __forceinline__ v16h load_frag(const _Float16* p) {
    union { v16h v; float4 f[2]; } u;
    u.f[0] = *(const float4*)(p);
    u.f[1] = *(const float4*)(p + 16);
    return u.v;
}

__device__ __forceinline__ v8f wmma_f16(v16h a, v16h b, v8f c) {
    return __builtin_amdgcn_wmma_f32_16x16x32_f16(false, a, false, b,
                                                  (short)0, c, false, false);
}

// CDNA5 async global->LDS copy (16B per lane), tracked by ASYNCcnt.
__device__ __forceinline__ void async_copy_b128(unsigned lds_off, const _Float16* g) {
    asm volatile("global_load_async_to_lds_b128 %0, %1, off"
                 :: "v"(lds_off), "v"(g) : "memory");
}
__device__ __forceinline__ void wait_async0() {
    asm volatile("s_wait_asynccnt 0x0" ::: "memory");
}

// ---------------------------------------------------------------------------
// Block-tiled GEMM core: 256 threads, 8 waves.
// Block computes a 256x64 output tile: wave wv owns rows [bm+wv*16, +16) and
// [bm+128+wv*16, +16), all 64 columns (4 n-tiles).
// Weight panel (64 x K, row-major (N,K)) is streamed in K-chunks of 32 into a
// double-buffered LDS tile via async copies; A fragments come from global.
// sW: __shared__ _Float16[2*64*32].
// ---------------------------------------------------------------------------
template <int KSTEPS>
__device__ __forceinline__ void gemm_core(const _Float16* __restrict__ A, int lda,
                                          const _Float16* __restrict__ W, int ldw,
                                          int bm, int n0, _Float16* sW,
                                          v8f acc[2][4]) {
    const int tid  = threadIdx.x;
    const int lane = tid & 31, wv = tid >> 5;
    const int lr   = lane & 15, kb = (lane >> 4) * 8;
    const int srow = tid >> 2;           // 0..63 : weight row this thread copies
    const int sseg = (tid & 3) * 8;      // 8-half segment within 32-half chunk row

    const unsigned ldsbase = (unsigned)(size_t)(void*)sW;
    const _Float16* wsrc = W + (size_t)(n0 + srow) * ldw + sseg;

    // prologue: chunk 0 -> buffer 0
    async_copy_b128(ldsbase + (unsigned)((srow * 32 + sseg) * 2), wsrc);

    const _Float16* apA = A + (size_t)(bm + wv * 16 + lr) * lda + kb;
    const _Float16* apB = A + (size_t)(bm + 128 + wv * 16 + lr) * lda + kb;

    for (int k = 0; k < KSTEPS; k++) {
        wait_async0();          // my slice of chunk k arrived
        __syncthreads();        // everyone's slice arrived; prev buffer free
        if (k + 1 < KSTEPS) {
            unsigned boff = ((k + 1) & 1) ? (unsigned)(64 * 32 * 2) : 0u;
            async_copy_b128(ldsbase + boff + (unsigned)((srow * 32 + sseg) * 2),
                            wsrc + (k + 1) * 32);
        }
        const int k0 = k * 32;
        v16h a0 = load_frag(apA + k0);
        v16h a1 = load_frag(apB + k0);
        const _Float16* bbase = sW + ((k & 1) ? 64 * 32 : 0);
#pragma unroll
        for (int nt = 0; nt < 4; nt++) {
            v16h b = load_frag(bbase + (nt * 16 + lr) * 32 + kb);
            acc[0][nt] = wmma_f16(a0, b, acc[0][nt]);
            acc[1][nt] = wmma_f16(a1, b, acc[1][nt]);
        }
    }
}

// ---------------------------------------------------------------------------
// Utility kernels
// ---------------------------------------------------------------------------
__global__ void k_f32_to_f16(const float* __restrict__ s, _Float16* __restrict__ d, int n) {
    int i = blockIdx.x * 256 + threadIdx.x;
    if (i < n) d[i] = (_Float16)s[i];
}

// x (B,C,H,W) -> t (B*HW, C)
__global__ void k_nchw_to_nc(const float* __restrict__ x, float* __restrict__ t) {
    size_t i = (size_t)blockIdx.x * 256 + threadIdx.x;
    int c   = (int)(i & 255);
    size_t r = i >> 8;
    int b_  = (int)(r / HW_);
    int p   = (int)(r % HW_);
    t[i] = x[((size_t)b_ * CH + c) * HW_ + p];
}

// t (B*HW, C) -> out (B,C,H,W)
__global__ void k_nc_to_nchw(const float* __restrict__ t, float* __restrict__ out) {
    size_t i = (size_t)blockIdx.x * 256 + threadIdx.x;
    int p   = (int)(i % HW_);
    size_t r = i / HW_;
    int c   = (int)(r & 255);
    int b_  = (int)(r >> 8);
    out[i] = t[((size_t)b_ * HW_ + p) * CH + c];
}

// ---------------------------------------------------------------------------
// LayerNorm (one wave per 256-channel token row)
// ---------------------------------------------------------------------------
__device__ __forceinline__ void ln_row(const float* __restrict__ ip,
                                       const float* __restrict__ g,
                                       const float* __restrict__ be,
                                       _Float16* __restrict__ op, int lane) {
    float4 x0 = *(const float4*)(ip);
    float4 x1 = *(const float4*)(ip + 4);
    float v[8] = {x0.x, x0.y, x0.z, x0.w, x1.x, x1.y, x1.z, x1.w};
    float s = 0.f, ss = 0.f;
#pragma unroll
    for (int j = 0; j < 8; j++) { s += v[j]; ss += v[j] * v[j]; }
#pragma unroll
    for (int m = 16; m >= 1; m >>= 1) {
        s  += __shfl_xor(s,  m, 32);
        ss += __shfl_xor(ss, m, 32);
    }
    float mu   = s * (1.f / 256.f);
    float var  = ss * (1.f / 256.f) - mu * mu;
    float rstd = rsqrtf(var + LN_EPS);
    _Float16 o[8];
#pragma unroll
    for (int j = 0; j < 8; j++) {
        int c = lane * 8 + j;
        o[j] = (_Float16)((v[j] - mu) * rstd * g[c] + be[c]);
    }
    *(float4*)op = *(const float4*)o;
}

// LN1: t -> windowed f16 (NWIN, 64, 256), rows 49..63 zeroed.
__global__ void k_ln_win(const float* __restrict__ t, const float* __restrict__ g,
                         const float* __restrict__ be, _Float16* __restrict__ out) {
    int win  = blockIdx.x;
    int wv   = threadIdx.x >> 5, lane = threadIdx.x & 31;
    int b_   = win >> 8, rem = win & 255, ph = rem >> 4, pw = rem & 15;
    for (int r = wv; r < WPAD; r += 8) {
        _Float16* op = out + ((size_t)win * WPAD + r) * CH + lane * 8;
        if (r < 49) {
            int hh = ph * WS + r / WS;
            int wc = pw * WS + r % WS;
            const float* ip = t + ((size_t)b_ * HW_ + hh * WW_ + wc) * CH + lane * 8;
            ln_row(ip, g, be, op, lane);
        } else {
            float4 z = {0.f, 0.f, 0.f, 0.f};
            *(float4*)op = z;
        }
    }
}

// LN2: t -> f16 (NTOK, 256), token order.
__global__ void k_ln2(const float* __restrict__ t, const float* __restrict__ g,
                      const float* __restrict__ be, _Float16* __restrict__ out) {
    int wv = threadIdx.x >> 5, lane = threadIdx.x & 31;
    int row = blockIdx.x * 8 + wv;
    ln_row(t + (size_t)row * CH + lane * 8, g, be,
           out + (size_t)row * CH + lane * 8, lane);
}

// ---------------------------------------------------------------------------
// GEMM kernels (block = 256x64 tile, async double-buffered weights)
// ---------------------------------------------------------------------------

// QKV: (MWIN x 256) @ (768 x 256)^T + bias -> f16 (MWIN x 768)
__global__ __launch_bounds__(256) void
k_gemm_qkv(const _Float16* __restrict__ A, const _Float16* __restrict__ W,
           const float* __restrict__ bias, _Float16* __restrict__ out) {
    __shared__ __align__(16) _Float16 sW[2 * 64 * 32];
    const int nB = 768 / 64;
    int bm = (blockIdx.x / nB) * 256, n0 = (blockIdx.x % nB) * 64;
    v8f acc[2][4] = {};
    gemm_core<8>(A, CH, W, CH, bm, n0, sW, acc);
    int lane = threadIdx.x & 31, wv = threadIdx.x >> 5;
    int lr = lane & 15, ro = (lane >> 4) * 8;
#pragma unroll
    for (int mt = 0; mt < 2; mt++)
#pragma unroll
        for (int nt = 0; nt < 4; nt++) {
            int col = n0 + nt * 16 + lr;
            float b = bias[col];
#pragma unroll
            for (int v = 0; v < 8; v++) {
                int row = bm + mt * 128 + wv * 16 + ro + v;
                out[(size_t)row * 768 + col] = (_Float16)(acc[mt][nt][v] + b);
            }
        }
}

// Proj: (MWIN x 256) @ (256 x 256)^T + bias, de-window, residual add into t.
__global__ __launch_bounds__(256) void
k_gemm_proj(const _Float16* __restrict__ A, const _Float16* __restrict__ W,
            const float* __restrict__ bias, float* __restrict__ t) {
    __shared__ __align__(16) _Float16 sW[2 * 64 * 32];
    const int nB = 256 / 64;
    int bm = (blockIdx.x / nB) * 256, n0 = (blockIdx.x % nB) * 64;
    v8f acc[2][4] = {};
    gemm_core<8>(A, CH, W, CH, bm, n0, sW, acc);
    int lane = threadIdx.x & 31, wv = threadIdx.x >> 5;
    int lr = lane & 15, ro = (lane >> 4) * 8;
#pragma unroll
    for (int mt = 0; mt < 2; mt++)
#pragma unroll
        for (int v = 0; v < 8; v++) {
            int mrow = bm + mt * 128 + wv * 16 + ro + v;
            int win = mrow >> 6, r = mrow & 63;
            if (r >= 49) continue;
            int b_ = win >> 8, rem = win & 255, ph = rem >> 4, pw = rem & 15;
            int hh = ph * WS + r / WS;
            int wc = pw * WS + r % WS;
            size_t ti = ((size_t)b_ * HW_ + hh * WW_ + wc) * CH;
#pragma unroll
            for (int nt = 0; nt < 4; nt++) {
                int col = n0 + nt * 16 + lr;
                t[ti + col] += acc[mt][nt][v] + bias[col];
            }
        }
}

// FFN conv1 (1x1): (NTOK x 256) @ (1024 x 256)^T -> bias+BN+GELU -> f16
__global__ __launch_bounds__(256) void
k_gemm_ffn1(const _Float16* __restrict__ A, const _Float16* __restrict__ W,
            const float* __restrict__ bias, const float* __restrict__ bng,
            const float* __restrict__ bnb, _Float16* __restrict__ out) {
    __shared__ __align__(16) _Float16 sW[2 * 64 * 32];
    const int nB = HID / 64;
    int bm = (blockIdx.x / nB) * 256, n0 = (blockIdx.x % nB) * 64;
    v8f acc[2][4] = {};
    gemm_core<8>(A, CH, W, CH, bm, n0, sW, acc);
    int lane = threadIdx.x & 31, wv = threadIdx.x >> 5;
    int lr = lane & 15, ro = (lane >> 4) * 8;
    const float rs = rsqrtf(1.0f + BN_EPS);
#pragma unroll
    for (int mt = 0; mt < 2; mt++)
#pragma unroll
        for (int nt = 0; nt < 4; nt++) {
            int col = n0 + nt * 16 + lr;
            float b = bias[col], sc = bng[col] * rs, sh = bnb[col];
#pragma unroll
            for (int v = 0; v < 8; v++) {
                int row = bm + mt * 128 + wv * 16 + ro + v;
                float x = (acc[mt][nt][v] + b) * sc + sh;
                out[(size_t)row * HID + col] = (_Float16)gelu_exact(x);
            }
        }
}

// FFN conv3 (1x1): (NTOK x 1024) @ (256 x 1024)^T -> bias+BN+GELU -> t +=
__global__ __launch_bounds__(256) void
k_gemm_ffn3(const _Float16* __restrict__ A, const _Float16* __restrict__ W,
            const float* __restrict__ bias, const float* __restrict__ bng,
            const float* __restrict__ bnb, float* __restrict__ t) {
    __shared__ __align__(16) _Float16 sW[2 * 64 * 32];
    const int nB = 256 / 64;
    int bm = (blockIdx.x / nB) * 256, n0 = (blockIdx.x % nB) * 64;
    v8f acc[2][4] = {};
    gemm_core<32>(A, HID, W, HID, bm, n0, sW, acc);
    int lane = threadIdx.x & 31, wv = threadIdx.x >> 5;
    int lr = lane & 15, ro = (lane >> 4) * 8;
    const float rs = rsqrtf(1.0f + BN_EPS);
#pragma unroll
    for (int mt = 0; mt < 2; mt++)
#pragma unroll
        for (int nt = 0; nt < 4; nt++) {
            int col = n0 + nt * 16 + lr;
            float b = bias[col], sc = bng[col] * rs, sh = bnb[col];
#pragma unroll
            for (int v = 0; v < 8; v++) {
                int row = bm + mt * 128 + wv * 16 + ro + v;
                float x = (acc[mt][nt][v] + b) * sc + sh;
                t[(size_t)row * CH + col] += gelu_exact(x);
            }
        }
}

// ---------------------------------------------------------------------------
// Attention: per (window, head) workgroup of 128 threads (4 waves).
// ---------------------------------------------------------------------------
__global__ void k_attn(const _Float16* __restrict__ qkv, const float* __restrict__ rpe,
                       _Float16* __restrict__ obuf) {
    __shared__ __align__(16) _Float16 sV[HD * WPAD];    // V^T: (dim, token)
    __shared__ __align__(16) float    sS[WPAD * WPAD];  // scores
    __shared__ __align__(16) _Float16 sP[WPAD * WPAD];  // probs f16

    int win  = blockIdx.x >> 3;
    int h    = blockIdx.x & 7;
    int tid  = threadIdx.x;
    int lane = tid & 31, wv = tid >> 5;
    int lr   = lane & 15, kb = (lane >> 4) * 8, ro = (lane >> 4) * 8;

    const _Float16* base = qkv + (size_t)win * WPAD * 768;
    const _Float16* Q    = base + h * HD;
    const _Float16* Km   = base + CH + h * HD;
    const _Float16* Vm   = base + 2 * CH + h * HD;

    {   // stage V transposed
        int tok = tid >> 1;
        int d0  = (tid & 1) * 16;
        const _Float16* vp = Vm + (size_t)tok * 768 + d0;
#pragma unroll
        for (int j = 0; j < 16; j++) sV[(d0 + j) * WPAD + tok] = vp[j];
    }
    __syncthreads();

    {   // S = Q @ K^T
        int m0 = wv * 16;
        v16h a = load_frag(Q + (size_t)(m0 + lr) * 768 + kb);
#pragma unroll
        for (int tn = 0; tn < 4; tn++) {
            int n0 = tn * 16;
            v16h b = load_frag(Km + (size_t)(n0 + lr) * 768 + kb);
            v8f c = {};
            c = wmma_f16(a, b, c);
#pragma unroll
            for (int v = 0; v < 8; v++)
                sS[(m0 + ro + v) * WPAD + n0 + lr] = c[v];
        }
    }
    __syncthreads();

    if (tid < WPAD) {   // softmax + RPE bias (scale = 1/sqrt(32))
        int q = tid;
        if (q < 49) {
            const float scale = 0.17677669529663688f;
            int qx = q / WS, qy = q % WS;
            float mx = -1e30f;
            for (int k = 0; k < 49; k++) {
                int kx = k / WS, ky = k % WS;
                int ridx = (qx - kx + WS - 1) * (2 * WS - 1) + (qy - ky + WS - 1);
                float s = sS[q * WPAD + k] * scale + rpe[ridx * NH + h];
                sS[q * WPAD + k] = s;
                mx = fmaxf(mx, s);
            }
            float sum = 0.f;
            for (int k = 0; k < 49; k++) {
                float e = __expf(sS[q * WPAD + k] - mx);
                sS[q * WPAD + k] = e;
                sum += e;
            }
            float inv = 1.f / sum;
            for (int k = 0; k < 49; k++)
                sP[q * WPAD + k] = (_Float16)(sS[q * WPAD + k] * inv);
            for (int k = 49; k < WPAD; k++) sP[q * WPAD + k] = (_Float16)0.f;
        } else {
            for (int k = 0; k < WPAD; k++) sP[q * WPAD + k] = (_Float16)0.f;
        }
    }
    __syncthreads();

    {   // O = P @ V
        int m0 = wv * 16;
        v8f acc[2] = {};
        for (int k0 = 0; k0 < WPAD; k0 += 32) {
            v16h a = load_frag(sP + (m0 + lr) * WPAD + k0 + kb);
#pragma unroll
            for (int nt = 0; nt < 2; nt++) {
                v16h b = load_frag(sV + (nt * 16 + lr) * WPAD + k0 + kb);
                acc[nt] = wmma_f16(a, b, acc[nt]);
            }
        }
#pragma unroll
        for (int nt = 0; nt < 2; nt++)
#pragma unroll
            for (int v = 0; v < 8; v++) {
                int q = m0 + ro + v;
                obuf[((size_t)win * WPAD + q) * CH + h * HD + nt * 16 + lr] =
                    (_Float16)acc[nt][v];
            }
    }
}

// ---------------------------------------------------------------------------
// Depthwise 3x3, channel-last (B*HW, 1024), + bias + BN + GELU.
// ---------------------------------------------------------------------------
__global__ void k_dw(const _Float16* __restrict__ h1, const float* __restrict__ w,
                     const float* __restrict__ bias, const float* __restrict__ bng,
                     const float* __restrict__ bnb, _Float16* __restrict__ h2) {
    int pix = blockIdx.x;
    int b_ = pix / HW_, p = pix % HW_;
    int y = p / WW_, x = p % WW_;
    int c0 = threadIdx.x * 4;
    float acc[4] = {0.f, 0.f, 0.f, 0.f};
#pragma unroll
    for (int dy = -1; dy <= 1; dy++) {
        int yy = y + dy;
        if ((unsigned)yy >= HH_) continue;
#pragma unroll
        for (int dx = -1; dx <= 1; dx++) {
            int xx = x + dx;
            if ((unsigned)xx >= WW_) continue;
            const _Float16* sp = h1 + ((size_t)b_ * HW_ + yy * WW_ + xx) * HID + c0;
            int widx = (dy + 1) * 3 + (dx + 1);
#pragma unroll
            for (int j = 0; j < 4; j++)
                acc[j] += (float)sp[j] * w[(c0 + j) * 9 + widx];
        }
    }
    const float rs = rsqrtf(1.0f + BN_EPS);
    _Float16* op = h2 + ((size_t)b_ * HW_ + p) * HID + c0;
#pragma unroll
    for (int j = 0; j < 4; j++) {
        int c = c0 + j;
        float v = (acc[j] + bias[c]) * (bng[c] * rs) + bnb[c];
        op[j] = (_Float16)gelu_exact(v);
    }
}

// ---------------------------------------------------------------------------
// Host launch
// ---------------------------------------------------------------------------
extern "C" void kernel_launch(void* const* d_in, const int* in_sizes, int n_in,
                              void* d_out, int out_size, void* d_ws, size_t ws_size,
                              hipStream_t stream) {
    const float* x      = (const float*)d_in[0];
    const float* ln1_g  = (const float*)d_in[1];
    const float* ln1_b  = (const float*)d_in[2];
    const float* qkv_w  = (const float*)d_in[3];
    const float* qkv_b  = (const float*)d_in[4];
    const float* rpe    = (const float*)d_in[5];
    const float* proj_w = (const float*)d_in[6];
    const float* proj_b = (const float*)d_in[7];
    const float* ln2_g  = (const float*)d_in[8];
    const float* ln2_b  = (const float*)d_in[9];
    const float* c1_w   = (const float*)d_in[10];
    const float* c1_b   = (const float*)d_in[11];
    const float* bn1_g  = (const float*)d_in[12];
    const float* bn1_b  = (const float*)d_in[13];
    const float* dw_w   = (const float*)d_in[14];
    const float* dw_b   = (const float*)d_in[15];
    const float* bn2_g  = (const float*)d_in[16];
    const float* bn2_b  = (const float*)d_in[17];
    const float* c3_w   = (const float*)d_in[18];
    const float* c3_b   = (const float*)d_in[19];
    const float* bn3_g  = (const float*)d_in[20];
    const float* bn3_b  = (const float*)d_in[21];

    // Workspace layout
    const size_t T_BYTES = (size_t)NTOK * CH * 4;    // residual stream (f32)
    const size_t A_BYTES = (size_t)NTOK * HID * 2;   // ln_win / h1
    const size_t B_BYTES = (size_t)NTOK * HID * 2;   // qkv / h2
    const size_t C_BYTES = (size_t)MWIN * CH * 2;    // obuf / ln2 out
    char* ws = (char*)d_ws;
    float*     t    = (float*)ws;
    _Float16*  bufA = (_Float16*)(ws + T_BYTES);
    _Float16*  bufB = (_Float16*)(ws + T_BYTES + A_BYTES);
    _Float16*  bufC = (_Float16*)(ws + T_BYTES + A_BYTES + B_BYTES);
    _Float16*  w16  = (_Float16*)(ws + T_BYTES + A_BYTES + B_BYTES + C_BYTES);

    const size_t QKVW = 768 * 256, PROJW = 256 * 256, C1W = 1024 * 256, C3W = 256 * 1024;
    const size_t WBLK = QKVW + PROJW + C1W + C3W;
    for (int i = 0; i < 2; i++) {
        _Float16* wb = w16 + i * WBLK;
        k_f32_to_f16<<<(QKVW + 255) / 256, 256, 0, stream>>>(qkv_w  + i * QKVW, wb, (int)QKVW);
        k_f32_to_f16<<<(PROJW + 255) / 256, 256, 0, stream>>>(proj_w + i * PROJW, wb + QKVW, (int)PROJW);
        k_f32_to_f16<<<(C1W  + 255) / 256, 256, 0, stream>>>(c1_w   + i * C1W, wb + QKVW + PROJW, (int)C1W);
        k_f32_to_f16<<<(C3W  + 255) / 256, 256, 0, stream>>>(c3_w   + i * C3W, wb + QKVW + PROJW + C1W, (int)C3W);
    }

    k_nchw_to_nc<<<(NTOK * CH) / 256, 256, 0, stream>>>(x, t);

    for (int i = 0; i < 2; i++) {
        _Float16* wb = w16 + i * WBLK;
        _Float16* qkvw16  = wb;
        _Float16* projw16 = wb + QKVW;
        _Float16* c1w16   = wb + QKVW + PROJW;
        _Float16* c3w16   = wb + QKVW + PROJW + C1W;

        // --- windowed MHSA branch ---
        k_ln_win<<<NWIN, 256, 0, stream>>>(t, ln1_g + i * CH, ln1_b + i * CH, bufA);
        // M=65536 -> 256 m-blocks, 12 n-blocks
        k_gemm_qkv<<<256 * 12, 256, 0, stream>>>(bufA, qkvw16, qkv_b + i * 768, bufB);
        k_attn<<<NWIN * NH, 128, 0, stream>>>(bufB, rpe + i * 169 * NH, bufC);
        // M=65536 -> 256 m-blocks, 4 n-blocks
        k_gemm_proj<<<256 * 4, 256, 0, stream>>>(bufC, projw16, proj_b + i * CH, t);

        // --- DW-CNN FFN branch ---
        k_ln2<<<NTOK / 8, 256, 0, stream>>>(t, ln2_g + i * CH, ln2_b + i * CH, bufC);
        // M=50176 -> 196 m-blocks, 16 n-blocks
        k_gemm_ffn1<<<196 * 16, 256, 0, stream>>>(bufC, c1w16, c1_b + i * HID,
                                                  bn1_g + i * HID, bn1_b + i * HID, bufA);
        k_dw<<<NTOK, 256, 0, stream>>>(bufA, dw_w + i * HID * 9, dw_b + i * HID,
                                       bn2_g + i * HID, bn2_b + i * HID, bufB);
        // M=50176 -> 196 m-blocks, 4 n-blocks
        k_gemm_ffn3<<<196 * 4, 256, 0, stream>>>(bufB, c3w16, c3_b + i * CH,
                                                 bn3_g + i * CH, bn3_b + i * CH, t);
    }

    k_nc_to_nchw<<<(NTOK * CH) / 256, 256, 0, stream>>>(t, (float*)d_out);
}
